// ESFF_62070867361835
// MI455X (gfx1250) — matrix operand order repaired and checked by reading, not compiled, not run on hardware
//
#include <hip/hip_runtime.h>
#include <hip/hip_bf16.h>

typedef __attribute__((ext_vector_type(16))) _Float16 v16h;
typedef __attribute__((ext_vector_type(8)))  _Float16 v8h;
typedef __attribute__((ext_vector_type(8)))  float    v8f;

// ---------------------------------------------------------------------------
// Generic batched WMMA GEMM:  C[ob,ib] = scale * (A[ob,ib] @ B[ob,ib]) + bias
//   A: [M,K] f16, row stride lda (halves), batch strides a_ob / a_ib
//   B: element (n,k) at B[n*bn + k]  (k contiguous; weight-[N,K] layout)
//   K is a compile-time constant (multiple of 32) -> fully unrolled K loop.
//   Edge handling: A rows / B cols are CLAMPED to the last valid row/col
//   (duplicated data only affects output elements the store mask drops),
//   so all fragment loads are unconditional b128 pairs. Stores are masked
//   per-lane only in edge tiles (wave-uniform branch).
// Fragment layouts per CDNA5 ISA 7.12.2 (wave32):
//   A  lane<16: row=lane,  elems = K{kb..kb+7, kb+16..kb+23}
//      lane>=16: row=lane-16, elems = K{kb+8..kb+15, kb+24..kb+31}
//   B  lane<16: col=lane,  elems = K{kb..kb+7, kb+16..kb+23}; lane>=16: K{kb+8..kb+15, kb+24..kb+31}
//   C/D VGPR j: row = j + 8*(lane>=16), col = lane&15
// ---------------------------------------------------------------------------
template <int K>
__global__ void k_wmma_gemm(const _Float16* __restrict__ A, int lda, long a_ob, long a_ib, int M,
                            const _Float16* __restrict__ B, int bn, long b_ob, long b_ib, int N,
                            float scale, const float* __restrict__ bias,
                            float* __restrict__ Cf, _Float16* __restrict__ Ch,
                            int ldc, long c_ob, long c_ib,
                            int Mtiles, int Ntiles)
{
  int wave = blockIdx.x * (blockDim.x >> 5) + (threadIdx.x >> 5);
  if (wave >= Mtiles * Ntiles) return;            // wave-uniform exit
  int lane = threadIdx.x & 31;
  int mt = wave / Ntiles;
  int nt = wave - mt * Ntiles;
  int ob = blockIdx.z, ib = blockIdx.y;

  const _Float16* Ab = A + (long)ob * a_ob + (long)ib * a_ib;
  const _Float16* Bb = B + (long)ob * b_ob + (long)ib * b_ib;

  int r  = lane & 15;
  int hi = lane >> 4;
  int arow = min(mt * 16 + r, M - 1);   // clamp: dup rows only feed masked outputs
  int bcol = min(nt * 16 + r, N - 1);   // clamp: dup cols only feed masked outputs

  const _Float16* ap = Ab + (long)arow * lda + hi * 8;
  const _Float16* bp = Bb + (long)bcol * bn + hi * 8;

  v8f acc = {};
#pragma unroll
  for (int kb = 0; kb < K; kb += 32) {
    v16h af, bf;
    const v8h* pa = (const v8h*)(ap + kb);
    v8h alo = pa[0];           // K{kb+hi*8 .. +7}
    v8h aup = pa[2];           // K{kb+16+hi*8 .. +7}
    const v8h* pb = (const v8h*)(bp + kb);
    v8h b0 = pb[0];            // K{kb+hi*8 .. +7}
    v8h b1 = pb[2];            // K{kb+16+hi*8 .. +7}
#pragma unroll
    for (int j = 0; j < 8; ++j) {
      af[j] = alo[j];
      af[j + 8] = aup[j];
      bf[j] = b0[j];
      bf[j + 8] = b1[j];
    }
    acc = __builtin_amdgcn_wmma_f32_16x16x32_f16(false, af, false, bf,
                                                 (short)0, acc, false, false);
  }

  float*    Cfb = Cf ? Cf + (long)ob * c_ob + (long)ib * c_ib : (float*)0;
  _Float16* Chb = Ch ? Ch + (long)ob * c_ob + (long)ib * c_ib : (_Float16*)0;
  int col = nt * 16 + r;
  float badd = bias ? bias[bcol] : 0.f;
  bool full = (mt * 16 + 16 <= M) && (nt * 16 + 16 <= N);   // wave-uniform
  if (full) {
#pragma unroll
    for (int j = 0; j < 8; ++j) {
      int row = mt * 16 + j + hi * 8;
      float v = acc[j] * scale + badd;
      if (Cfb) Cfb[(long)row * ldc + col] = v;
      if (Chb) Chb[(long)row * ldc + col] = (_Float16)v;
    }
  } else {
#pragma unroll
    for (int j = 0; j < 8; ++j) {
      int row = mt * 16 + j + hi * 8;
      if (row < M && col < N) {
        float v = acc[j] * scale + badd;
        if (Cfb) Cfb[(long)row * ldc + col] = v;
        if (Chb) Chb[(long)row * ldc + col] = (_Float16)v;
      }
    }
  }
}

template <int K>
static void launch_gemm(hipStream_t stream,
                        const _Float16* A, int lda, long a_ob, long a_ib, int M,
                        const _Float16* B, int bn, long b_ob, long b_ib, int N,
                        float scale, const float* bias,
                        float* Cf, _Float16* Ch, int ldc, long c_ob, long c_ib,
                        int nb_in, int nb_out)
{
  int Mt = (M + 15) / 16, Nt = (N + 15) / 16;
  dim3 grid((Mt * Nt + 7) / 8, nb_in, nb_out);
  k_wmma_gemm<K><<<grid, dim3(256), 0, stream>>>(A, lda, a_ob, a_ib, M,
                                                 B, bn, b_ob, b_ib, N,
                                                 scale, bias,
                                                 Cf, Ch, ldc, c_ob, c_ib, Mt, Nt);
}

// ---------------------------------------------------------------------------
// Small support kernels
// ---------------------------------------------------------------------------
__global__ void k_f32_to_f16(const float* __restrict__ s, _Float16* __restrict__ d, int n) {
  int i = blockIdx.x * blockDim.x + threadIdx.x;
  if (i < n) d[i] = (_Float16)s[i];
}

// x[28,196,256] -> mean over groups of 7 -> xm[4,196,256] (f32 + f16)
__global__ void k_mean7(const float* __restrict__ x, float* __restrict__ xm,
                        _Float16* __restrict__ xmh) {
  int i = blockIdx.x * blockDim.x + threadIdx.x;
  if (i >= 4 * 196 * 256) return;
  int b  = i / (196 * 256);
  int pc = i - b * (196 * 256);
  float s = 0.f;
#pragma unroll
  for (int g = 0; g < 7; ++g) s += x[(long)(b * 7 + g) * (196 * 256) + pc];
  float v = s * (1.0f / 7.0f);
  xm[i]  = v;
  xmh[i] = (_Float16)v;
}

// 2x2 avg pool: xm[4,14,14,256] -> xs[4,7,7,256] (f16)
__global__ void k_pool2(const float* __restrict__ xm, _Float16* __restrict__ xsh) {
  int i = blockIdx.x * blockDim.x + threadIdx.x;
  if (i >= 4 * 49 * 256) return;
  int b = i / (49 * 256);
  int r = i - b * (49 * 256);
  int q = r / 256, c = r - q * 256;
  int qh = q / 7, qw = q - qh * 7;
  const float* base = xm + (long)b * 196 * 256;
  float s = base[((2 * qh) * 14 + 2 * qw) * 256 + c]
          + base[((2 * qh) * 14 + 2 * qw + 1) * 256 + c]
          + base[((2 * qh + 1) * 14 + 2 * qw) * 256 + c]
          + base[((2 * qh + 1) * 14 + 2 * qw + 1) * 256 + c];
  xsh[i] = (_Float16)(s * 0.25f);
}

// LAE dilated 3x3 window attention. qkv: [4*196, 768] f32,
// channel o = which*256 + dil*128 + head*32 + d. Output f16 [4*196, 256],
// channel = dil*128 + head*32 + d. Zero-padded borders: logit 0, v 0.
__global__ void k_lae_dilate(const float* __restrict__ qkv, _Float16* __restrict__ out_h) {
  int t = blockIdx.x * blockDim.x + threadIdx.x;
  if (t >= 4 * 196 * 2 * 4) return;
  int head = t & 3;
  int dil  = (t >> 2) & 1;
  int p    = (t >> 3) % 196;
  int b    = t / (196 * 8);
  int h = p / 14, w = p - h * 14;
  int d = dil ? 3 : 1;
  const float* qb = qkv + (long)(b * 196 + p) * 768 + dil * 128 + head * 32;

  float logit[9];
  int   pn[9];
#pragma unroll
  for (int i = 0; i < 3; ++i)
#pragma unroll
    for (int j = 0; j < 3; ++j) {
      int idx = i * 3 + j;
      int hh = h + (i - 1) * d, ww = w + (j - 1) * d;
      if (hh >= 0 && hh < 14 && ww >= 0 && ww < 14) {
        int pj = hh * 14 + ww;
        pn[idx] = pj;
        const float* kb = qkv + (long)(b * 196 + pj) * 768 + 256 + dil * 128 + head * 32;
        float s = 0.f;
        for (int c = 0; c < 32; ++c) s += qb[c] * kb[c];
        logit[idx] = s * 0.17677669529663687f;   // 32^-0.5
      } else {
        pn[idx] = -1;
        logit[idx] = 0.f;                         // zero-padded k -> logit 0
      }
    }
  float mx = logit[0];
  for (int i = 1; i < 9; ++i) mx = fmaxf(mx, logit[i]);
  float wgt[9], sum = 0.f;
  for (int i = 0; i < 9; ++i) { wgt[i] = expf(logit[i] - mx); sum += wgt[i]; }
  float inv = 1.f / sum;

  _Float16* o = out_h + (long)(b * 196 + p) * 256 + dil * 128 + head * 32;
  for (int c = 0; c < 32; ++c) {
    float acc = 0.f;
    for (int i = 0; i < 9; ++i)
      if (pn[i] >= 0)
        acc += wgt[i] * qkv[(long)(b * 196 + pn[i]) * 768 + 512 + dil * 128 + head * 32 + c];
    o[c] = (_Float16)(acc * inv);
  }
}

// GAE softmax over 49 keys; logits f32 [16*196, 49] (already scaled 0.177),
// probs f16 [16*196, 64] zero-padded.
__global__ void k_gae_softmax(const float* __restrict__ logits, _Float16* __restrict__ probs) {
  int t = blockIdx.x * blockDim.x + threadIdx.x;
  if (t >= 16 * 196) return;
  const float* l = logits + (long)t * 49;
  _Float16*    o = probs + (long)t * 64;
  float mx = l[0];
  for (int i = 1; i < 49; ++i) mx = fmaxf(mx, l[i]);
  float e[49], sum = 0.f;
  for (int i = 0; i < 49; ++i) { e[i] = expf(l[i] - mx); sum += e[i]; }
  float inv = 1.f / sum;
  for (int i = 0; i < 49; ++i) o[i] = (_Float16)(e[i] * inv);
  for (int i = 49; i < 64; ++i) o[i] = (_Float16)0.f;
}

// kv_h [196,256] (v = cols 128..255) -> vT [b,head,32,64] f16, ks padded 49..63
__global__ void k_transpose_v(const _Float16* __restrict__ kv_h, _Float16* __restrict__ vT) {
  int i = blockIdx.x * blockDim.x + threadIdx.x;
  if (i >= 4 * 4 * 32 * 64) return;
  int ks   = i & 63;
  int dd   = (i >> 6) & 31;
  int head = (i >> 11) & 3;
  int b    = i >> 13;
  _Float16 v = (_Float16)0.f;
  if (ks < 49) v = kv_h[(long)(b * 49 + ks) * 256 + 128 + head * 32 + dd];
  vT[i] = v;
}

// Classifier: out[b,o] = dot(cls_w[o], concat(lae[b],gae[b]) per-pixel) + cls_b[o]
__global__ void k_classifier(const float* __restrict__ lae, const float* __restrict__ gae,
                             const float* __restrict__ cls_w, const float* __restrict__ cls_b,
                             float* __restrict__ out) {
  int b = blockIdx.x >> 1;
  int o = blockIdx.x & 1;
  const float* wrow = cls_w + (long)o * 75264;
  float acc = 0.f;
  for (int i = threadIdx.x; i < 75264; i += blockDim.x) {
    int p = i / 384, c = i - p * 384;
    float v = (c < 256) ? lae[(long)(b * 196 + p) * 256 + c]
                        : gae[(long)(b * 196 + p) * 128 + (c - 256)];
    acc += v * wrow[i];
  }
  __shared__ float red[256];
  red[threadIdx.x] = acc;
  __syncthreads();
  for (int s = 128; s > 0; s >>= 1) {
    if (threadIdx.x < s) red[threadIdx.x] += red[threadIdx.x + s];
    __syncthreads();
  }
  if (threadIdx.x == 0) out[b * 2 + o] = red[0] + cls_b[o];
}

// ---------------------------------------------------------------------------
extern "C" void kernel_launch(void* const* d_in, const int* in_sizes, int n_in,
                              void* d_out, int out_size, void* d_ws, size_t ws_size,
                              hipStream_t stream) {
  const float* x          = (const float*)d_in[0];
  const float* qkv_w      = (const float*)d_in[1];
  const float* lae_proj_w = (const float*)d_in[2];
  const float* lae_proj_b = (const float*)d_in[3];
  const float* lq_w       = (const float*)d_in[4];
  const float* lkv_w      = (const float*)d_in[5];
  const float* lproj_w    = (const float*)d_in[6];
  const float* lproj_b    = (const float*)d_in[7];
  const float* cls_w      = (const float*)d_in[8];
  const float* cls_b      = (const float*)d_in[9];
  float* out = (float*)d_out;

  char* ws = (char*)d_ws;
  size_t off = 0;
  auto take = [&](size_t bytes) -> char* {
    off = (off + 255) & ~(size_t)255;
    char* p = ws + off;
    off += bytes;
    return p;
  };
  float*    xm_f      = (float*)   take((size_t)4 * 196 * 256 * 4);
  _Float16* xm_h      = (_Float16*)take((size_t)4 * 196 * 256 * 2);
  _Float16* xs_h      = (_Float16*)take((size_t)4 * 49 * 256 * 2);
  _Float16* qkvw_h    = (_Float16*)take((size_t)768 * 256 * 2);
  _Float16* laepw_h   = (_Float16*)take((size_t)256 * 256 * 2);
  _Float16* lqw_h     = (_Float16*)take((size_t)128 * 256 * 2);
  _Float16* lkvw_h    = (_Float16*)take((size_t)256 * 256 * 2);
  _Float16* lprojw_h  = (_Float16*)take((size_t)128 * 128 * 2);
  float*    qkv_f     = (float*)   take((size_t)784 * 768 * 4);
  _Float16* lae_att_h = (_Float16*)take((size_t)784 * 256 * 2);
  float*    lae_out_f = (float*)   take((size_t)784 * 256 * 4);
  _Float16* q_h       = (_Float16*)take((size_t)784 * 128 * 2);
  _Float16* kv_h      = (_Float16*)take((size_t)196 * 256 * 2);
  float*    attn_f    = (float*)   take((size_t)16 * 196 * 49 * 4);
  _Float16* attn_h    = (_Float16*)take((size_t)16 * 196 * 64 * 2);
  _Float16* vT_h      = (_Float16*)take((size_t)16 * 32 * 64 * 2);
  _Float16* gae_pre_h = (_Float16*)take((size_t)784 * 128 * 2);
  float*    gae_out_f = (float*)   take((size_t)784 * 128 * 4);
  (void)ws_size; (void)in_sizes; (void)n_in; (void)out_size;

  // Weight f32 -> f16 conversions
  k_f32_to_f16<<<(768 * 256 + 255) / 256, 256, 0, stream>>>(qkv_w, qkvw_h, 768 * 256);
  k_f32_to_f16<<<(256 * 256 + 255) / 256, 256, 0, stream>>>(lae_proj_w, laepw_h, 256 * 256);
  k_f32_to_f16<<<(128 * 256 + 255) / 256, 256, 0, stream>>>(lq_w, lqw_h, 128 * 256);
  k_f32_to_f16<<<(256 * 256 + 255) / 256, 256, 0, stream>>>(lkv_w, lkvw_h, 256 * 256);
  k_f32_to_f16<<<(128 * 128 + 255) / 256, 256, 0, stream>>>(lproj_w, lprojw_h, 128 * 128);

  // Batch-mean and pool
  k_mean7<<<(4 * 196 * 256 + 255) / 256, 256, 0, stream>>>(x, xm_f, xm_h);
  k_pool2<<<(4 * 49 * 256 + 255) / 256, 256, 0, stream>>>(xm_f, xs_h);

  // QKV GEMM: [784,256] x weight[768,256] -> qkv_f [784,768]
  launch_gemm<256>(stream, xm_h, 256, 0, 0, 784,
                   qkvw_h, 256, 0, 0, 768,
                   1.f, nullptr, qkv_f, nullptr, 768, 0, 0, 1, 1);

  // LAE dilated attention (scalar) -> lae_att_h [784,256]
  k_lae_dilate<<<(4 * 196 * 8 + 255) / 256, 256, 0, stream>>>(qkv_f, lae_att_h);

  // LAE projection: [784,256] x [256,256] + bias -> lae_out_f
  launch_gemm<256>(stream, lae_att_h, 256, 0, 0, 784,
                   laepw_h, 256, 0, 0, 256,
                   1.f, lae_proj_b, lae_out_f, nullptr, 256, 0, 0, 1, 1);

  // GAE q: [784,256] x [128,256] -> q_h [784,128]
  launch_gemm<256>(stream, xm_h, 256, 0, 0, 784,
                   lqw_h, 256, 0, 0, 128,
                   1.f, nullptr, nullptr, q_h, 128, 0, 0, 1, 1);

  // GAE kv: [196,256] x [256,256] -> kv_h [196,256] (k = cols 0..127, v = 128..255)
  launch_gemm<256>(stream, xs_h, 256, 0, 0, 196,
                   lkvw_h, 256, 0, 0, 256,
                   1.f, nullptr, nullptr, kv_h, 256, 0, 0, 1, 1);

  // Batched QK^T (grid.y = head, grid.z = b): [196,32] x k[49,32] * 0.177 -> attn_f
  launch_gemm<32>(stream, q_h, 128, (long)196 * 128, 32, 196,
                  kv_h, 256, (long)49 * 256, 32, 49,
                  0.177f, nullptr, attn_f, nullptr, 49,
                  (long)4 * 196 * 49, (long)196 * 49, 4, 4);

  // Softmax over 49 -> attn_h [16*196, 64] (zero-padded)
  k_gae_softmax<<<(16 * 196 + 255) / 256, 256, 0, stream>>>(attn_f, attn_h);

  // V transpose: kv_h v-half -> vT [16, 32, 64] zero-padded
  k_transpose_v<<<(16 * 32 * 64 + 255) / 256, 256, 0, stream>>>(kv_h, vT_h);

  // Batched attn @ V: [196,64] x vT[32,64] -> gae_pre_h [784,128]
  launch_gemm<64>(stream, attn_h, 64, (long)4 * 196 * 64, (long)196 * 64, 196,
                  vT_h, 64, (long)4 * 32 * 64, (long)32 * 64, 32,
                  1.f, nullptr, nullptr, gae_pre_h, 128,
                  (long)196 * 128, 32, 4, 4);

  // GAE projection: [784,128] x [128,128] + bias -> gae_out_f
  launch_gemm<128>(stream, gae_pre_h, 128, 0, 0, 784,
                   lprojw_h, 128, 0, 0, 128,
                   1.f, lproj_b, gae_out_f, nullptr, 128, 0, 0, 1, 1);

  // Classifier -> out [4,2]
  k_classifier<<<8, 256, 0, stream>>>(lae_out_f, gae_out_f, cls_w, cls_b, out);
}
